// LSTM_CIFG_9320079033018
// MI455X (gfx1250) — matrix operand order repaired and checked by reading, not compiled, not run on hardware
//
#include <hip/hip_runtime.h>
#include <hip/hip_bf16.h>

// ---------------------------------------------------------------------------
// CIFG-LSTM cell, MI455X (gfx1250, wave32, WMMA + async global->LDS copies).
//
// Phase 0 (one-time ~77MB): f32 -> bf16 conversion of x, h_prev and the six
// 1024x1024 weights into d_ws (Wx{i,c,o} / Wh{i,c,o} packed [3][1024][1024]).
//
// Phase 1 (compute-bound, AI ~ 570 FLOP/B): fused K=2048 loop, 3 gate
// accumulator sets per wave (A tile reused 3x). Staging uses CDNA5
// GLOBAL_LOAD_ASYNC_TO_LDS_B128 (ASYNCcnt) so no VGPRs are consumed or
// spilled holding in-flight data; weights are L2-resident (12.6MB << 192MB L2).
// ---------------------------------------------------------------------------

#ifdef __has_builtin
# if __has_builtin(__builtin_amdgcn_global_load_async_to_lds_b128)
#  define HAVE_ASYNC_COPY 1
# endif
# if __has_builtin(__builtin_amdgcn_s_wait_asynccnt)
#  define HAVE_ASYNC_WAIT 1
# endif
#endif

typedef __attribute__((ext_vector_type(16))) __bf16 v16bf;
typedef __attribute__((ext_vector_type(8)))  float  v8f;
typedef __attribute__((ext_vector_type(4)))  int    v4i;

typedef __attribute__((address_space(1))) v4i* gv4i_ptr;   // global
typedef __attribute__((address_space(3))) v4i* lv4i_ptr;   // LDS

#define BATCH 16384
#define HID   1024
#define KTOT  2048
#define BM    128
#define BN    64
#define BK    32
#define NSTEP (KTOT / BK)
#define KP    40        // padded LDS k-stride (bf16): 80B, avoids bank conflicts

// d_ws layout (bf16 elements)
#define WS_XB   ((size_t)0)                    // x        [16384][1024]
#define WS_HB   ((size_t)16777216)             // h_prev   [16384][1024]
#define WS_WX   ((size_t)33554432)             // Wx{i,c,o} [3][1024][1024]
#define WS_WH   ((size_t)36700160)             // Wh{i,c,o} [3][1024][1024]
#define WMAT    ((size_t)1048576)              // 1024*1024

union Frag16 { v16bf v; uint4 q[2]; };

__device__ __forceinline__ unsigned f2bf(float f) {
    unsigned u = __float_as_uint(f);
    return (u + 0x7FFFu + ((u >> 16) & 1u)) >> 16;     // round-to-nearest-even
}
__device__ __forceinline__ unsigned pk2(float a, float b) {
    return f2bf(a) | (f2bf(b) << 16);
}
__device__ __forceinline__ float fsigmoid(float x) {
    return 1.0f / (1.0f + __expf(-x));
}
__device__ __forceinline__ float ftanh(float x) {
    return 1.0f - 2.0f / (1.0f + __expf(2.0f * x));
}

// 16B global -> LDS copy: async (no VGPR payload) when available
__device__ __forceinline__ void cp16(const unsigned short* g, unsigned short* l) {
#ifdef HAVE_ASYNC_COPY
    __builtin_amdgcn_global_load_async_to_lds_b128(
        (gv4i_ptr)g, (lv4i_ptr)l, 0, 0);
#else
    *(uint4*)l = *(const uint4*)g;
#endif
}
__device__ __forceinline__ void cp_wait() {
#ifdef HAVE_ASYNC_COPY
# ifdef HAVE_ASYNC_WAIT
    __builtin_amdgcn_s_wait_asynccnt(0);
# else
    asm volatile("s_wait_asynccnt 0x0" ::: "memory");
# endif
#endif
}

// ---- one-time f32 -> bf16 conversion, 8 elements / thread -----------------
__global__ void cvt_f32_bf16(const float* __restrict__ src,
                             unsigned short* __restrict__ dst, int n8) {
    int i = blockIdx.x * blockDim.x + threadIdx.x;
    if (i >= n8) return;
    const float4* s = (const float4*)src + 2 * (size_t)i;
    float4 a = s[0], b = s[1];
    uint4 p;
    p.x = pk2(a.x, a.y); p.y = pk2(a.z, a.w);
    p.z = pk2(b.x, b.y); p.w = pk2(b.z, b.w);
    ((uint4*)dst)[i] = p;
}

// ---- main fused GEMM + CIFG epilogue --------------------------------------
__global__ __launch_bounds__(256) __attribute__((amdgpu_waves_per_eu(2)))
void lstm_cifg_wmma(const unsigned short* __restrict__ xb,
                    const unsigned short* __restrict__ hb,
                    const unsigned short* __restrict__ wxb,   // [3][1024][1024]
                    const unsigned short* __restrict__ whb,   // [3][1024][1024]
                    const float* __restrict__ cprev,
                    const float* __restrict__ bxi, const float* __restrict__ bhi,
                    const float* __restrict__ bxc, const float* __restrict__ bhc,
                    const float* __restrict__ bxo, const float* __restrict__ bho,
                    float* __restrict__ out)
{
    __shared__ __align__(16) unsigned short As[2][BM][KP];        // 20480 B
    __shared__ __align__(16) unsigned short Bs[2][3][BN][KP];     // 30720 B

    const int tid   = threadIdx.x;
    const int lane  = tid & 31;
    const int wave  = tid >> 5;
    const int waveM = wave >> 1;         // 0..3
    const int waveN = wave & 1;          // 0..1
    const int mlo   = lane & 15;
    const int khalf = lane >> 4;

    const int rowA0 = blockIdx.x * BM;
    const int colB0 = blockIdx.y * BN;

    // staging geometry: thread owns A rows (tid>>2, +64) and B row tid>>2 of
    // each gate, one 16B chunk at k-offset (tid&3)*8
    const int kc = (tid & 3) * 8;
    const int r0 = tid >> 2;             // 0..63

    v8f acc[3][2][2] = {};

    // issue 5 global->LDS copies for k-step s into buffer buf (no VGPR payload)
    auto stage = [&](int s, int buf) {
        const int k0 = s * BK;
        const unsigned short* Ab = (k0 < 1024) ? xb  : hb;
        const unsigned short* Bb = (k0 < 1024) ? wxb : whb;
        const int kk = (k0 & 1023) + kc;
        cp16(Ab + (size_t)(rowA0 + r0)      * 1024 + kk, &As[buf][r0][kc]);
        cp16(Ab + (size_t)(rowA0 + r0 + 64) * 1024 + kk, &As[buf][r0 + 64][kc]);
        #pragma unroll
        for (int g = 0; g < 3; ++g)
            cp16(Bb + g * WMAT + (size_t)(colB0 + r0) * 1024 + kk,
                 &Bs[buf][g][r0][kc]);
    };

    auto compute = [&](int buf) {
        Frag16 fa[2], fb[3][2];
        // fetch ALL fragments first (16 ds_load_b128), then the WMMA burst
        #pragma unroll
        for (int mt = 0; mt < 2; ++mt) {
            // A 16x32 bf16: lanes 0-15 K 0-7/16-23, lanes 16-31 K 8-15/24-31
            const int M  = waveM * 32 + mt * 16 + mlo;
            const int kb = khalf * 8;
            fa[mt].q[0] = *(const uint4*)&As[buf][M][kb];
            fa[mt].q[1] = *(const uint4*)&As[buf][M][kb + 16];
        }
        #pragma unroll
        for (int g = 0; g < 3; ++g) {
            #pragma unroll
            for (int nt = 0; nt < 2; ++nt) {
                // B 32x16 bf16: lanes 0-15 K 0-7/16-23, lanes 16-31 K 8-15/24-31
                const int N  = waveN * 32 + nt * 16 + mlo;
                const int kb = khalf * 8;
                fb[g][nt].q[0] = *(const uint4*)&Bs[buf][g][N][kb];
                fb[g][nt].q[1] = *(const uint4*)&Bs[buf][g][N][kb + 16];
            }
        }
        #pragma unroll
        for (int g = 0; g < 3; ++g)
            #pragma unroll
            for (int nt = 0; nt < 2; ++nt)
                #pragma unroll
                for (int mt = 0; mt < 2; ++mt)
                    acc[g][mt][nt] = __builtin_amdgcn_wmma_f32_16x16x32_bf16(
                        false, fa[mt].v, false, fb[g][nt].v,
                        (short)0, acc[g][mt][nt], false, false);
    };

    // ---- pipelined main loop, double-buffered LDS -------------------------
    int buf = 0;
    stage(0, 0);
    cp_wait();
    __syncthreads();
    #pragma unroll 1
    for (int s = 0; s < NSTEP; ++s) {
        if (s + 1 < NSTEP) stage(s + 1, buf ^ 1);   // async, overlaps compute
        compute(buf);
        cp_wait();                                   // our copies for buf^1 done
        __syncthreads();
        buf ^= 1;
    }

    // ---- epilogue: bias + CIFG nonlinearity -------------------------------
    #pragma unroll
    for (int nt = 0; nt < 2; ++nt) {
        const int col = colB0 + waveN * 32 + nt * 16 + mlo;
        const float bi = bxi[col] + bhi[col];
        const float bc = bxc[col] + bhc[col];
        const float bo = bxo[col] + bho[col];
        #pragma unroll
        for (int mt = 0; mt < 2; ++mt) {
            #pragma unroll
            for (int r = 0; r < 8; ++r) {
                const int row = rowA0 + waveM * 32 + mt * 16 + khalf * 8 + r;
                const size_t idx = (size_t)row * HID + col;
                const float pi = acc[0][mt][nt][r] + bi;
                const float pc = acc[1][mt][nt][r] + bc;
                const float po = acc[2][mt][nt][r] + bo;
                const float ig = fsigmoid(pi);
                const float cn = ftanh(pc) * ig + (1.0f - ig) * cprev[idx];
                const float hn = fsigmoid(po) * ftanh(cn);
                out[idx] = hn;                                   // h_next
                out[(size_t)BATCH * HID + idx] = cn;             // c_next
            }
        }
    }
}

extern "C" void kernel_launch(void* const* d_in, const int* in_sizes, int n_in,
                              void* d_out, int out_size, void* d_ws, size_t ws_size,
                              hipStream_t stream) {
    const float* x     = (const float*)d_in[0];
    const float* hprev = (const float*)d_in[1];
    const float* cprev = (const float*)d_in[2];
    const float* Wxi   = (const float*)d_in[3];
    const float* bxi   = (const float*)d_in[4];
    const float* Whi   = (const float*)d_in[5];
    const float* bhi   = (const float*)d_in[6];
    const float* Wxc   = (const float*)d_in[7];
    const float* bxc   = (const float*)d_in[8];
    const float* Whc   = (const float*)d_in[9];
    const float* bhc   = (const float*)d_in[10];
    const float* Wxo   = (const float*)d_in[11];
    const float* bxo   = (const float*)d_in[12];
    const float* Who   = (const float*)d_in[13];
    const float* bho   = (const float*)d_in[14];
    float* out = (float*)d_out;

    unsigned short* ws = (unsigned short*)d_ws;

    // --- one-time f32 -> bf16 conversion into workspace ---
    const int n8_act = (BATCH * HID) / 8;      // 2097152
    const int n8_w   = (HID * HID) / 8;        // 131072
    dim3 cb(256);
    cvt_f32_bf16<<<dim3((n8_act + 255) / 256), cb, 0, stream>>>(x,     ws + WS_XB, n8_act);
    cvt_f32_bf16<<<dim3((n8_act + 255) / 256), cb, 0, stream>>>(hprev, ws + WS_HB, n8_act);
    cvt_f32_bf16<<<dim3((n8_w + 255) / 256), cb, 0, stream>>>(Wxi, ws + WS_WX + 0 * WMAT, n8_w);
    cvt_f32_bf16<<<dim3((n8_w + 255) / 256), cb, 0, stream>>>(Wxc, ws + WS_WX + 1 * WMAT, n8_w);
    cvt_f32_bf16<<<dim3((n8_w + 255) / 256), cb, 0, stream>>>(Wxo, ws + WS_WX + 2 * WMAT, n8_w);
    cvt_f32_bf16<<<dim3((n8_w + 255) / 256), cb, 0, stream>>>(Whi, ws + WS_WH + 0 * WMAT, n8_w);
    cvt_f32_bf16<<<dim3((n8_w + 255) / 256), cb, 0, stream>>>(Whc, ws + WS_WH + 1 * WMAT, n8_w);
    cvt_f32_bf16<<<dim3((n8_w + 255) / 256), cb, 0, stream>>>(Who, ws + WS_WH + 2 * WMAT, n8_w);

    // --- fused 3-gate GEMM + CIFG epilogue ---
    dim3 grid(BATCH / BM, HID / BN, 1);   // 128 x 16
    dim3 block(256, 1, 1);                // 8 wave32s
    lstm_cifg_wmma<<<grid, block, 0, stream>>>(
        ws + WS_XB, ws + WS_HB, ws + WS_WX, ws + WS_WH,
        cprev, bxi, bhi, bxc, bhc, bxo, bho, out);
}
